// TemporalFusion_module_25451976196132
// MI455X (gfx1250) — matrix operand order repaired and checked by reading, not compiled, not run on hardware
//
#include <hip/hip_runtime.h>
#include <hip/hip_bf16.h>
#include <math.h>

// Problem constants (reference: N=4, C=32, H=W=256)
#define Cc  32
#define Nn  4
#define Hh  256
#define Ww  256
#define PP  65536      // Hh*Ww
#define NP  262144     // Nn*PP

typedef __attribute__((ext_vector_type(16))) _Float16 v16h;
typedef __attribute__((ext_vector_type(8)))  float    v8f;

__device__ __forceinline__ int imin(int a, int b) { return a < b ? a : b; }
__device__ __forceinline__ int imax(int a, int b) { return a > b ? a : b; }

// D(16x16,f32) = A(16x32,f16) x B(32x16,f16) + C  -- full K=32 reduction per instruction
__device__ __forceinline__ v8f wmma_f16(v16h a, v16h b, v8f c) {
  return __builtin_amdgcn_wmma_f32_16x16x32_f16(
      /*neg_a=*/false, a, /*neg_b=*/false, b,
      /*c_mod=*/(short)0, c, /*reuse_a=*/false, /*reuse_b=*/false);
}

// ISA 7.12.2: 16-bit A 16x32. Lane L: row m=L&15, half h=L>>4.
// this code puts K = 16h + j in element j (matches B; not the ISA table order)
__device__ __forceinline__ int a_kidx(int j, int h) {
  return 16 * h + j;
}
// B 32x16: lane L: col n=L&15, half j holds K = 16h + j
// C/D 16x16 f32: vgpr r: M = r + 8h, N = L&15

__device__ __forceinline__ float actf(float v, int ACT) {
  if (ACT == 1) return fmaxf(v, 0.f);
  if (ACT == 2) return 1.f / (1.f + expf(-v));
  return v;
}

// ---------------------------------------------------------------------------
// LayerNorm over channel dim (C=32), NCHW, per spatial position
// ---------------------------------------------------------------------------
__global__ void __launch_bounds__(256)
k_layernorm(const float* __restrict__ X, const float* __restrict__ g,
            const float* __restrict__ b, float* __restrict__ Y) {
  int q = blockIdx.x * blockDim.x + threadIdx.x;     // grid covers NP exactly
  int n = q >> 16, s = q & (PP - 1);
  const float* xp = X + (size_t)n * Cc * PP + s;
  float v[Cc], mu = 0.f;
#pragma unroll
  for (int c = 0; c < Cc; ++c) { v[c] = xp[(size_t)c * PP]; mu += v[c]; }
  mu *= (1.f / Cc);
  float var = 0.f;
#pragma unroll
  for (int c = 0; c < Cc; ++c) { float d = v[c] - mu; var += d * d; }
  float rs = rsqrtf(var * (1.f / Cc) + 1e-5f);
  float* yp = Y + (size_t)n * Cc * PP + s;
#pragma unroll
  for (int c = 0; c < Cc; ++c) yp[(size_t)c * PP] = (v[c] - mu) * rs * g[c] + b[c];
}

// ---------------------------------------------------------------------------
// 1x1 conv as WMMA GEMM, fully compile-time specialized.
// Y[o,p] = act( sum_c W[o,c] X[c,p] + bias[o] ) (+R[o,p])
// Wave: 16 positions; CIN in k-chunks of 32; COUT/16 o-tiles.
// ACT: 0=none 1=relu 2=sigmoid
// ---------------------------------------------------------------------------
template <int CIN, int COUT, int ACT, bool HAS_RES>
__global__ void __launch_bounds__(256)
k_conv1x1(const float* __restrict__ X, const float* __restrict__ W,
          const float* __restrict__ Bias, const float* __restrict__ R,
          float* __restrict__ Y) {
  constexpr int NT = (COUT + 15) / 16;
  constexpr int NK = (CIN + 31) / 32;

  int gtid = blockIdx.x * blockDim.x + threadIdx.x;
  int wave = gtid >> 5;
  int lane = threadIdx.x & 31;
  int m = lane & 15, h = lane >> 4;
  int p = wave * 16 + m;                 // this lane's position (B/D column)
  int n = p >> 16, s = p & (PP - 1);
  const float* xb = X + (size_t)n * CIN * PP + s;

  v8f acc[NT] = {};

#pragma unroll
  for (int kc = 0; kc < NK; ++kc) {
    const int kb = kc * 32;
    // B fragment: unconditional clamped loads + value select (no EXEC branches)
    v16h bf;
#pragma unroll
    for (int j = 0; j < 16; ++j) {
      int c = kb + 16 * h + j;
      float v = xb[(size_t)imin(c, CIN - 1) * PP];   // always in-bounds
      if (CIN % 32 != 0) v = (c < CIN) ? v : 0.f;    // v_cndmask, no branch
      bf[j] = (_Float16)v;
    }
#pragma unroll
    for (int t = 0; t < NT; ++t) {
      int o = t * 16 + m;
      v16h af;
#pragma unroll
      for (int j = 0; j < 16; ++j) {
        int c = kb + a_kidx(j, h);
        float w = W[o * CIN + imin(c, CIN - 1)];
        if (CIN % 32 != 0) w = (c < CIN) ? w : 0.f;
        af[j] = (_Float16)w;
      }
      acc[t] = wmma_f16(af, bf, acc[t]);
    }
  }
#pragma unroll
  for (int t = 0; t < NT; ++t) {
#pragma unroll
    for (int r = 0; r < 8; ++r) {
      int o = t * 16 + r + 8 * h;
      float v = actf(acc[t][r] + Bias[o], ACT);
      size_t oi = (size_t)n * COUT * PP + (size_t)o * PP + s;
      if (HAS_RES) v += R[oi];
      Y[oi] = v;
    }
  }
}

// ---------------------------------------------------------------------------
// Dense 3x3 conv (pad=1), Cin=32, as 9 shifted WMMA GEMMs; COUT in {9,18}.
// Tiles: 16 contiguous x within one row -> shifted tiles stay in-row.
// ---------------------------------------------------------------------------
template <int COUT, int ACT>
__global__ void __launch_bounds__(256)
k_conv3x3(const float* __restrict__ X, const float* __restrict__ W,
          const float* __restrict__ Bias, float* __restrict__ Y) {
  constexpr int NT = (COUT + 15) / 16;

  int gtid = blockIdx.x * blockDim.x + threadIdx.x;
  int wave = gtid >> 5;
  int lane = threadIdx.x & 31;
  int m = lane & 15, h = lane >> 4;
  int p = wave * 16 + m;
  int n = p >> 16, s = p & (PP - 1);
  int y = s >> 8, x = s & (Ww - 1);
  const float* xb = X + (size_t)n * Cc * PP;

  v8f acc[NT] = {};

#pragma unroll
  for (int k = 0; k < 9; ++k) {
    const int ky = k / 3 - 1, kx = k % 3 - 1;
    int yy = y + ky, xx = x + kx;
    bool ok = (yy >= 0) && (yy < Hh) && (xx >= 0) && (xx < Ww);
    int sp = imin(imax(yy, 0), Hh - 1) * Ww + imin(imax(xx, 0), Ww - 1);  // in-bounds
    v16h bf;
#pragma unroll
    for (int j = 0; j < 16; ++j) {
      int c = 16 * h + j;
      float v = xb[(size_t)c * PP + sp];             // unconditional load
      bf[j] = (_Float16)(ok ? v : 0.f);              // select
    }
#pragma unroll
    for (int t = 0; t < NT; ++t) {
      int o = t * 16 + m;
      v16h af;
#pragma unroll
      for (int j = 0; j < 16; ++j) {
        int c = a_kidx(j, h);
        float w = W[(imin(o, COUT - 1) * Cc + c) * 9 + k];  // clamped, in-bounds
        if (COUT % 16 != 0) w = (o < COUT) ? w : 0.f;       // select
        af[j] = (_Float16)w;
      }
      acc[t] = wmma_f16(af, bf, acc[t]);
    }
  }
#pragma unroll
  for (int t = 0; t < NT; ++t) {
#pragma unroll
    for (int r = 0; r < 8; ++r) {
      int o = t * 16 + r + 8 * h;
      if (o < COUT) {
        float v = actf(acc[t][r] + Bias[o], ACT);
        Y[(size_t)n * COUT * PP + (size_t)o * PP + s] = v;
      }
    }
  }
}

// ---------------------------------------------------------------------------
// Deformable conv: out[o,p] = sum_{k,c} W[o,c,k] * msk[k,p] * bilinear(x, k-offset)
// K = 9 taps x 32 channels = 288; 9 k-chunks x 2 o-tiles = 18 WMMAs / 16 positions.
// Mask + bilinear corner weights folded into the B fragment.
// ---------------------------------------------------------------------------
__global__ void __launch_bounds__(256)
k_deform(const float* __restrict__ Xf, const float* __restrict__ OFF,
         const float* __restrict__ MSK, const float* __restrict__ Wt,
         float* __restrict__ Y) {
  int gtid = blockIdx.x * blockDim.x + threadIdx.x;
  int wave = gtid >> 5;
  int lane = threadIdx.x & 31;
  int m = lane & 15, h = lane >> 4;
  int p = wave * 16 + m;
  int n = p >> 16, s = p & (PP - 1);
  int y = s >> 8, x = s & (Ww - 1);
  const float* xb = Xf + (size_t)n * Cc * PP;
  const float* ob = OFF + (size_t)n * 18 * PP + s;
  const float* mb = MSK + (size_t)n * 9 * PP + s;

  v8f acc0 = {}, acc1 = {};
#pragma unroll
  for (int k = 0; k < 9; ++k) {
    const int ky = k / 3 - 1, kx = k % 3 - 1;
    float dy = ob[(size_t)(2 * k) * PP];
    float dx = ob[(size_t)(2 * k + 1) * PP];
    float mk = mb[(size_t)k * PP];
    float ys = (float)(y + ky) + dy;
    float xs = (float)(x + kx) + dx;
    float fy = floorf(ys), fx = floorf(xs);
    float wy1 = ys - fy, wx1 = xs - fx;
    float wy0 = 1.f - wy1, wx0 = 1.f - wx1;
    int iy0 = (int)fy, ix0 = (int)fx, iy1 = iy0 + 1, ix1 = ix0 + 1;
    bool vy0 = (iy0 >= 0) && (iy0 < Hh), vy1 = (iy1 >= 0) && (iy1 < Hh);
    bool vx0 = (ix0 >= 0) && (ix0 < Ww), vx1 = (ix1 >= 0) && (ix1 < Ww);
    int cy0 = imin(imax(iy0, 0), Hh - 1), cy1 = imin(imax(iy1, 0), Hh - 1);
    int cx0 = imin(imax(ix0, 0), Ww - 1), cx1 = imin(imax(ix1, 0), Ww - 1);
    int s00 = cy0 * Ww + cx0, s01 = cy0 * Ww + cx1;
    int s10 = cy1 * Ww + cx0, s11 = cy1 * Ww + cx1;
    float w00 = (vy0 && vx0) ? wy0 * wx0 * mk : 0.f;
    float w01 = (vy0 && vx1) ? wy0 * wx1 * mk : 0.f;
    float w10 = (vy1 && vx0) ? wy1 * wx0 * mk : 0.f;
    float w11 = (vy1 && vx1) ? wy1 * wx1 * mk : 0.f;

    v16h bf;
#pragma unroll
    for (int j = 0; j < 16; ++j) {
      int c = 16 * h + j;
      const float* cp = xb + (size_t)c * PP;
      float v = cp[s00] * w00 + cp[s01] * w01 + cp[s10] * w10 + cp[s11] * w11;
      bf[j] = (_Float16)v;
    }
    v16h a0, a1;
#pragma unroll
    for (int j = 0; j < 16; ++j) {
      int c = a_kidx(j, h);
      a0[j] = (_Float16)Wt[((m) * Cc + c) * 9 + k];        // o-tile 0: o = m
      a1[j] = (_Float16)Wt[((16 + m) * Cc + c) * 9 + k];   // o-tile 1: o = 16+m
    }
    acc0 = wmma_f16(a0, bf, acc0);
    acc1 = wmma_f16(a1, bf, acc1);
  }
#pragma unroll
  for (int r = 0; r < 8; ++r) {
    int o0 = r + 8 * h, o1 = 16 + r + 8 * h;
    Y[(size_t)n * Cc * PP + (size_t)o0 * PP + s] = acc0[r];
    Y[(size_t)n * Cc * PP + (size_t)o1 * PP + s] = acc1[r];
  }
}

// ---------------------------------------------------------------------------
// Depthwise 3x3, pad=1 (groups == channels). One thread per output element.
// ---------------------------------------------------------------------------
template <int CCH>
__global__ void __launch_bounds__(256)
k_dw3x3(const float* __restrict__ X, const float* __restrict__ W,
        const float* __restrict__ Bias, float* __restrict__ Y) {
  int t = blockIdx.x * blockDim.x + threadIdx.x;   // over Nn*CCH*PP
  int x = t & (Ww - 1), y = (t >> 8) & (Hh - 1);
  int plane = t >> 16;                             // n*CCH + c
  int c = plane % CCH;
  int pb = plane << 16;
  float acc = Bias[c];
#pragma unroll
  for (int k = 0; k < 9; ++k) {
    const int ky = k / 3 - 1, kx = k % 3 - 1;
    int yy = y + ky, xx = x + kx;
    bool ok = (yy >= 0) && (yy < Hh) && (xx >= 0) && (xx < Ww);
    int sp = imin(imax(yy, 0), Hh - 1) * Ww + imin(imax(xx, 0), Ww - 1);
    float v = X[(size_t)pb + sp];                  // unconditional, in-bounds
    acc += W[c * 9 + k] * (ok ? v : 0.f);
  }
  Y[t] = acc;
}

// Fused depthwise 3x3 over implicit x_bi = concat(x_fc, x_fc*flowf, x_fc*flowb) (96 ch)
__global__ void __launch_bounds__(256)
k_dw3x3_tri(const float* __restrict__ Xf, const float* __restrict__ Ff,
            const float* __restrict__ Fb, const float* __restrict__ W,
            const float* __restrict__ Bias, float* __restrict__ Y) {
  int t = blockIdx.x * blockDim.x + threadIdx.x;   // over Nn*96*PP
  int x = t & (Ww - 1), y = (t >> 8) & (Hh - 1);
  int plane = t >> 16;                             // n*96 + c
  int c = plane % 96, n = plane / 96;
  int g = c >> 5, cc = c & 31;
  int pb = (n * Cc + cc) << 16;
  const float* F = (g == 1) ? Ff : Fb;             // used only when g != 0
  float acc = Bias[c];
#pragma unroll
  for (int k = 0; k < 9; ++k) {
    const int ky = k / 3 - 1, kx = k % 3 - 1;
    int yy = y + ky, xx = x + kx;
    bool ok = (yy >= 0) && (yy < Hh) && (xx >= 0) && (xx < Ww);
    int sp = imin(imax(yy, 0), Hh - 1) * Ww + imin(imax(xx, 0), Ww - 1);
    int off = pb + sp;
    float v = Xf[off];
    float f = F[off];
    if (g != 0) v *= f;
    acc += W[c * 9 + k] * (ok ? v : 0.f);
  }
  Y[t] = acc;
}

// GEGLU: a * 0.5*b*(1+erf(b/sqrt(2))), split of 96ch into 48+48
__global__ void __launch_bounds__(256)
k_geglu(const float* __restrict__ H96, float* __restrict__ G48) {
  int t = blockIdx.x * blockDim.x + threadIdx.x;   // over Nn*48*PP
  int s = t & (PP - 1);
  int u = t >> 16;                                 // n*48 + c
  int c = u % 48, n = u / 48;
  float a  = H96[((size_t)(n * 96 + c) << 16) + s];
  float bb = H96[((size_t)(n * 96 + 48 + c) << 16) + s];
  float gl = 0.5f * bb * (1.f + erff(bb * 0.70710678118654752f));
  G48[t] = a * gl;
}

// ---------------------------------------------------------------------------
// Launch: params appear in _make_params insertion order at d_in[3..50]
// ---------------------------------------------------------------------------
extern "C" void kernel_launch(void* const* d_in, const int* in_sizes, int n_in,
                              void* d_out, int out_size, void* d_ws, size_t ws_size,
                              hipStream_t stream) {
  const float* X  = (const float*)d_in[0];
  const float* FF = (const float*)d_in[1];
  const float* FB = (const float*)d_in[2];
#define PM(i) ((const float*)d_in[(i)])

  float* ws = (float*)d_ws;
  const size_t CSz = (size_t)Cc * NP;      // one 32-channel f32 tensor (8.39M elems)
  float* b0   = ws + 0 * CSz;              // xn      -> fwd_dc
  float* b1   = ws + 1 * CSz;              // ffn     -> back_dc
  float* b2   = ws + 2 * CSz;              // fbn     -> flowf
  float* b3   = ws + 3 * CSz;              // tmpC    -> flowb
  float* b4   = ws + 4 * CSz;              // x_fc (persists)
  float* offf = ws + 5 * CSz;              // 18*NP   (dead after deform)
  float* mskf = offf + (size_t)18 * NP;    // 9*NP
  float* offb = mskf + (size_t)9 * NP;     // 18*NP
  float* mskb = offb + (size_t)18 * NP;    // 9*NP   (ends < 7*CSz)
  float* t2a  = ws + 7 * CSz;              // 2C (gate ping)   lifetime: gates only
  float* t2b  = ws + 9 * CSz;              // 2C (gate pong)   total ws = 11*CSz
  float* h96  = ws + 5 * CSz;              // 3C, written after offf..mskb+t2a dead
  float* g48  = ws + 8 * CSz;              // 1.5C, written after t2a/t2b dead

  const int TB = 256;
  dim3 blk(TB);
  dim3 gLN(NP / TB);                       // 1024 blocks, 1 thread/pos
  dim3 gW((NP / 16) / (TB / 32));          // 2048 blocks, 1 wave per 16-pos tile

  // LayerNorms
  k_layernorm<<<gLN, blk, 0, stream>>>(X,  PM(3), PM(4), b0);
  k_layernorm<<<gLN, blk, 0, stream>>>(FF, PM(5), PM(6), b1);
  k_layernorm<<<gLN, blk, 0, stream>>>(FB, PM(7), PM(8), b2);

  // x_fc = dw3x3(relu(1x1(xn)))
  k_conv1x1<32, 32, 1, false><<<gW, blk, 0, stream>>>(b0, PM(9), PM(10), nullptr, b3);
  k_dw3x3<32><<<dim3((Nn * 32 * PP) / TB), blk, 0, stream>>>(b3, PM(11), PM(12), b4);

  // forward offsets / mask
  k_conv1x1<32, 32, 1, false><<<gW, blk, 0, stream>>>(b1, PM(13), PM(14), nullptr, b3);
  k_conv3x3<18, 0><<<gW, blk, 0, stream>>>(b3, PM(15), PM(16), offf);
  k_conv1x1<32, 32, 1, false><<<gW, blk, 0, stream>>>(b1, PM(17), PM(18), nullptr, b3);
  k_conv3x3<9, 2><<<gW, blk, 0, stream>>>(b3, PM(19), PM(20), mskf);
  // backward offsets / mask
  k_conv1x1<32, 32, 1, false><<<gW, blk, 0, stream>>>(b2, PM(22), PM(23), nullptr, b3);
  k_conv3x3<18, 0><<<gW, blk, 0, stream>>>(b3, PM(24), PM(25), offb);
  k_conv1x1<32, 32, 1, false><<<gW, blk, 0, stream>>>(b2, PM(26), PM(27), nullptr, b3);
  k_conv3x3<9, 2><<<gW, blk, 0, stream>>>(b3, PM(28), PM(29), mskb);

  // deformable convs
  k_deform<<<gW, blk, 0, stream>>>(b4, offf, mskf, PM(21), b0);   // fwd_dc
  k_deform<<<gW, blk, 0, stream>>>(b4, offb, mskb, PM(30), b1);   // back_dc

  // gate mf -> flowf (b2)
  k_conv1x1<32, 64, 0, false><<<gW, blk, 0, stream>>>(b0, PM(31), PM(32), nullptr, t2a);
  k_dw3x3<64><<<dim3((Nn * 64 * PP) / TB), blk, 0, stream>>>(t2a, PM(33), PM(34), t2b);
  k_conv1x1<64, 64, 1, false><<<gW, blk, 0, stream>>>(t2b, PM(35), PM(36), nullptr, t2a);
  k_conv1x1<64, 32, 2, false><<<gW, blk, 0, stream>>>(t2a, PM(37), PM(38), nullptr, b2);
  // gate mb -> flowb (b3)
  k_conv1x1<32, 64, 0, false><<<gW, blk, 0, stream>>>(b1, PM(39), PM(40), nullptr, t2a);
  k_dw3x3<64><<<dim3((Nn * 64 * PP) / TB), blk, 0, stream>>>(t2a, PM(41), PM(42), t2b);
  k_conv1x1<64, 64, 1, false><<<gW, blk, 0, stream>>>(t2b, PM(43), PM(44), nullptr, t2a);
  k_conv1x1<64, 32, 2, false><<<gW, blk, 0, stream>>>(t2a, PM(45), PM(46), nullptr, b3);

  // fl1 depthwise over implicit concat -> h96; GEGLU -> g48; fl2 + residual -> out
  k_dw3x3_tri<<<dim3((Nn * 96 * PP) / TB), blk, 0, stream>>>(b4, b2, b3, PM(47), PM(48), h96);
  k_geglu<<<dim3((Nn * 48 * PP) / TB), blk, 0, stream>>>(h96, g48);
  k_conv1x1<48, 32, 0, true><<<gW, blk, 0, stream>>>(g48, PM(49), PM(50), X, (float*)d_out);
#undef PM
}
